// BruteForceUpdater_23003844837552
// MI455X (gfx1250) — matrix operand order repaired and checked by reading, not compiled, not run on hardware
//
#include <hip/hip_runtime.h>
#include <hip/hip_bf16.h>

// Problem constants (from reference): IN=128, HID=64, NFW=8256, 2*NFW=16512, B=256
#define C_IN   128
#define C_HID  64
#define C_NFW  8256
#define C_NH   16512
#define C_B    256

typedef __attribute__((ext_vector_type(16))) __bf16 v16bf;
typedef __attribute__((ext_vector_type(8)))  __bf16 v8bf;
typedef __attribute__((ext_vector_type(8)))  float  v8f;
typedef __attribute__((ext_vector_type(4)))  float  v4f;

// ---- fragment helpers ------------------------------------------------------

static __device__ inline v8bf cvt8v(v4f a, v4f b) {
  v8bf r;
  r[0] = (__bf16)a[0]; r[1] = (__bf16)a[1]; r[2] = (__bf16)a[2]; r[3] = (__bf16)a[3];
  r[4] = (__bf16)b[0]; r[5] = (__bf16)b[1]; r[6] = (__bf16)b[2]; r[7] = (__bf16)b[3];
  return r;
}

static __device__ inline v8bf cvt8(const float* __restrict__ p) {
  return cvt8v(*(const v4f*)p, *(const v4f*)(p + 4));
}

static __device__ inline v16bf join16(v8bf lo, v8bf hi) {
  return __builtin_shufflevector(lo, hi, 0,1,2,3,4,5,6,7,8,9,10,11,12,13,14,15);
}

// A (16x32 bf16, M x K): lane<16 -> row mbase+lane, K = {0..7, 16..23};
//                        lane>=16 -> same rows, K = {8..15, 24..31}. fp32 src.
static __device__ inline v16bf load_a_f32(const float* __restrict__ src, int ld,
                                          int mbase, int kslab, int lane) {
  int lhi = (lane >> 4) & 1;
  const float* row = src + (size_t)(mbase + (lane & 15)) * ld + kslab + lhi * 8;
  return join16(cvt8(row), cvt8(row + 16));
}

// Same A layout, bf16 source (our V buffer).
static __device__ inline v16bf load_a_bf16(const __bf16* __restrict__ src, int ld,
                                           int mbase, int kslab, int lane) {
  int lhi = (lane >> 4) & 1;
  const __bf16* row = src + (size_t)(mbase + (lane & 15)) * ld + kslab + lhi * 8;
  v8bf c0 = *(const v8bf*)row;
  v8bf c1 = *(const v8bf*)(row + 16);
  return join16(c0, c1);
}

// B (32x16 bf16, K x N), column n from row n of a (rows x K) fp32 matrix
// (B = W.T): lane<16 -> col nbase+lane, K={0..7,16..23}; lane>=16 -> K={8..15,24..31}.
static __device__ inline v16bf load_b_f32(const float* __restrict__ w, int ld,
                                          int nbase, int kslab, int lane) {
  int lhi = (lane >> 4) & 1;
  const float* row = w + (size_t)(nbase + (lane & 15)) * ld + kslab + lhi * 8;
  return join16(cvt8(row), cvt8(row + 16));
}

// B fragment from LDS (bf16, row j major, padded stride).
static __device__ inline v16bf load_b_lds(const __bf16* lds, int stride,
                                          int nloc, int kloc, int lane) {
  int lhi = (lane >> 4) & 1;
  const __bf16* row = lds + (nloc + (lane & 15)) * stride + kloc + lhi * 8;
  v8bf c0 = *(const v8bf*)row;
  v8bf c1 = *(const v8bf*)(row + 16);
  return join16(c0, c1);
}

static __device__ inline v8f wmma_bf16(v16bf a, v16bf b, v8f c) {
  return __builtin_amdgcn_wmma_f32_16x16x32_bf16(false, a, false, b, (short)0, c,
                                                 false, false);
}

// ---- kernel 1: V = xs @ W1.T  (256 x 16512, bf16 out, f32 acc) -------------
// Grid 129 x 256 threads. Wave w owns an N-subtile of 16 cols, all 16 M-tiles.
__global__ __launch_bounds__(256) void k_gemm1(const float* __restrict__ xs,
                                               const float* __restrict__ W1,
                                               __bf16* __restrict__ V) {
  const int lane = threadIdx.x & 31;
  const int wave = threadIdx.x >> 5;
  const int nbase = blockIdx.x * 128 + wave * 16;

  const v8f vzero = {0.f, 0.f, 0.f, 0.f, 0.f, 0.f, 0.f, 0.f};
  v8f acc[16];
#pragma unroll
  for (int i = 0; i < 16; ++i) acc[i] = vzero;

#pragma unroll
  for (int s = 0; s < 4; ++s) {
    const int k0 = s * 32;
    v16bf bf = load_b_f32(W1, C_IN, nbase, k0, lane);
#pragma unroll
    for (int mt = 0; mt < 16; ++mt) {
      v16bf af = load_a_f32(xs, C_IN, mt * 16, k0, lane);
      acc[mt] = wmma_bf16(af, bf, acc[mt]);
    }
  }

  const int lhi = (lane >> 4) & 1;
  const int col = nbase + (lane & 15);
#pragma unroll
  for (int mt = 0; mt < 16; ++mt) {
#pragma unroll
    for (int r = 0; r < 8; ++r) {
      const int m = mt * 16 + r + lhi * 8;
      V[(size_t)m * C_NH + col] = (__bf16)acc[mt][r];
    }
  }
}

// ---- kernel 2: U = V @ W2.T  (256 x 8256, f32) -----------------------------
// The memory-bound kernel: streams W2 (545 MB) exactly once at HBM rate.
// Per workgroup: 64 W2 rows; K-slabs of 128 staged fp32->bf16 into LDS shared
// by all 8 waves, DOUBLE-BUFFERED: next slab's global loads are issued before
// the current slab's WMMAs, and waited on only at the next convert, so HBM
// latency hides behind matrix math. Wave w computes M-tiles {2w,2w+1} x 4
// N-tiles (8 f32 16x16 accumulators = 64 VGPRs).
#define LDSTR 136  // padded bf16 row stride in LDS
__global__ __launch_bounds__(256) void k_gemm2(const __bf16* __restrict__ V,
                                               const float* __restrict__ W2,
                                               float* __restrict__ U) {
  __shared__ __attribute__((aligned(16))) __bf16 sB[2][64 * LDSTR];

  const int lane = threadIdx.x & 31;
  const int wave = threadIdx.x >> 5;
  const int jbase = blockIdx.x * 64;

  const int tj = threadIdx.x >> 2;  // 0..63 : W2 row within tile
  const int tq = threadIdx.x & 3;   // 0..3  : 32-element k chunk

  const v8f vzero = {0.f, 0.f, 0.f, 0.f, 0.f, 0.f, 0.f, 0.f};
  v8f acc[2][4];
#pragma unroll
  for (int a = 0; a < 2; ++a)
#pragma unroll
    for (int b = 0; b < 4; ++b) acc[a][b] = vzero;

  // this thread's fixed staging source: W2[jbase+tj][tq*32 ...]
  const float* srcBase = W2 + (size_t)(jbase + tj) * C_NH + tq * 32;

  // preload slab 0 into registers (32 fp32 = 8 x b128)
  v4f regs[8];
#pragma unroll
  for (int u = 0; u < 8; ++u) regs[u] = *(const v4f*)(srcBase + u * 4);

  int p = 0;
  for (int kb = 0; kb < C_NH; kb += 128) {
    // convert the slab we already hold -> LDS buffer p (waits on its loads)
    {
      __bf16* dst = &sB[p][tj * LDSTR + tq * 32];
#pragma unroll
      for (int u = 0; u < 4; ++u) {
        v8bf v = cvt8v(regs[2 * u], regs[2 * u + 1]);
        *(v8bf*)(dst + u * 8) = v;
      }
    }
    // issue next slab's loads now; they complete during this slab's WMMAs
    if (kb + 128 < C_NH) {
      const float* src = srcBase + kb + 128;
#pragma unroll
      for (int u = 0; u < 8; ++u) regs[u] = *(const v4f*)(src + u * 4);
    }
    __syncthreads();

#pragma unroll
    for (int s = 0; s < 4; ++s) {
      const int ks = s * 32;
      v16bf a0 = load_a_bf16(V, C_NH, wave * 32, kb + ks, lane);
      v16bf a1 = load_a_bf16(V, C_NH, wave * 32 + 16, kb + ks, lane);
#pragma unroll
      for (int nt = 0; nt < 4; ++nt) {
        v16bf bf = load_b_lds(&sB[p][0], LDSTR, nt * 16, ks, lane);
        acc[0][nt] = wmma_bf16(a0, bf, acc[0][nt]);
        acc[1][nt] = wmma_bf16(a1, bf, acc[1][nt]);
      }
    }
    __syncthreads();
    p ^= 1;
  }

  const int lhi = (lane >> 4) & 1;
  const int lc = lane & 15;
#pragma unroll
  for (int mt = 0; mt < 2; ++mt) {
#pragma unroll
    for (int nt = 0; nt < 4; ++nt) {
#pragma unroll
      for (int r = 0; r < 8; ++r) {
        const int m = wave * 32 + mt * 16 + r + lhi * 8;
        const int j = jbase + nt * 16 + lc;
        U[(size_t)m * C_NFW + j] = acc[mt][nt][r];
      }
    }
  }
}

// ---- kernel 3: elementwise sigmoid recurrence + per-row dot ----------------
// fw_{i+1}[j] = sigmoid(10*(fw_i[j] + U[i][j] - 0.5)) is independent per j.
// Block r (of 64) owns w1 row r (128 fw elements, one per thread) + its w2
// element (thread 0). U/xs loads for step i+1 are issued before step i's
// reduction so L2 latency stays off the sequential chain.
__global__ __launch_bounds__(128) void k_scan(const float* __restrict__ U,
                                              const float* __restrict__ xs,
                                              const float* __restrict__ fw0,
                                              float* __restrict__ partial) {
  const int r = blockIdx.x;
  const int c = threadIdx.x;
  const int wave = c >> 5;
  const int lane = c & 31;
  __shared__ float sred[4];

  float fw = fw0[r * C_IN + c];
  float fw2 = (c == 0) ? fw0[C_HID * C_IN + r] : 0.f;
  const float GAIN = 10.f, SHIFT = 0.5f;

  float u = U[(size_t)0 * C_NFW + r * C_IN + c];
  float xi = xs[c];
  float u2 = (c == 0) ? U[(size_t)0 * C_NFW + C_HID * C_IN + r] : 0.f;

  for (int i = 0; i < C_B; ++i) {
    // prefetch next step's operands (independent of the recurrence)
    float u_n = 0.f, xi_n = 0.f, u2_n = 0.f;
    if (i + 1 < C_B) {
      u_n = U[(size_t)(i + 1) * C_NFW + r * C_IN + c];
      xi_n = xs[(i + 1) * C_IN + c];
      if (c == 0) u2_n = U[(size_t)(i + 1) * C_NFW + C_HID * C_IN + r];
    }

    const float z = GAIN * (fw + u - SHIFT);
    fw = 1.0f / (1.0f + __expf(-z));
    float v = fw * xi;

    // wave32 tree reduction
    v += __shfl_xor(v, 16, 32);
    v += __shfl_xor(v, 8, 32);
    v += __shfl_xor(v, 4, 32);
    v += __shfl_xor(v, 2, 32);
    v += __shfl_xor(v, 1, 32);
    if (lane == 0) sred[wave] = v;

    if (c == 0) {  // w2 element recurrence (off the reduction's critical path)
      const float z2 = GAIN * (fw2 + u2 - SHIFT);
      fw2 = 1.0f / (1.0f + __expf(-z2));
    }
    __syncthreads();
    if (c == 0) {
      float h = sred[0] + sred[1] + sred[2] + sred[3];
      h = (h > 0.f) ? h : 0.f;
      partial[i * C_HID + r] = fw2 * h;
    }
    __syncthreads();

    u = u_n; xi = xi_n; u2 = u2_n;
  }
}

// ---- kernel 4: preds[i] = sum_r partial[i][r] ------------------------------
__global__ __launch_bounds__(256) void k_reduce(const float* __restrict__ partial,
                                                float* __restrict__ out) {
  const int i = threadIdx.x;
  float s = 0.f;
#pragma unroll 8
  for (int r = 0; r < C_HID; ++r) s += partial[i * C_HID + r];
  out[i] = s;
}

// ---- launcher --------------------------------------------------------------
extern "C" void kernel_launch(void* const* d_in, const int* in_sizes, int n_in,
                              void* d_out, int out_size, void* d_ws, size_t ws_size,
                              hipStream_t stream) {
  const float* x   = (const float*)d_in[0];  // (256,1,128) == (256,128)
  const float* W1  = (const float*)d_in[1];  // (16512,128)
  const float* W2  = (const float*)d_in[2];  // (8256,16512)
  const float* fw0 = (const float*)d_in[3];  // (8256,)
  float* out = (float*)d_out;                // (256,1) flat

  char* ws = (char*)d_ws;
  float*  U       = (float*)ws;                                   // 256*8256 f32
  __bf16* V       = (__bf16*)(ws + (size_t)C_B * C_NFW * 4);      // 256*16512 bf16
  float*  partial = (float*)(ws + (size_t)C_B * C_NFW * 4
                                + (size_t)C_B * C_NH * 2);        // 256*64 f32

  k_gemm1<<<dim3(C_NH / 128), dim3(256), 0, stream>>>(x, W1, V);
  k_gemm2<<<dim3(C_NFW / 64), dim3(256), 0, stream>>>(V, W2, U);
  k_scan<<<dim3(C_HID), dim3(128), 0, stream>>>(U, x, fw0, partial);
  k_reduce<<<dim3(1), dim3(256), 0, stream>>>(partial, out);
}
